// DGCNN_65764539236864
// MI455X (gfx1250) — matrix operand order repaired and checked by reading, not compiled, not run on hardware
//
#include <hip/hip_runtime.h>

// ---------- types for WMMA ----------
typedef __attribute__((ext_vector_type(16))) _Float16 v16h;
typedef __attribute__((ext_vector_type(8)))  _Float16 h8;
typedef __attribute__((ext_vector_type(8)))  float    v8f;

union V16 { v16h v; h8 h[2]; };

constexpr int BATCH = 8;
constexpr int NPTS  = 1024;
constexpr int KNN   = 20;
#define BNS 0.9999950000374997f   // 1/sqrt(1+1e-5)

// ---------- f32 -> f16 convert with column zero-padding ----------
__global__ void cvt_pad_f16(const float* __restrict__ src, _Float16* __restrict__ dst,
                            int rows, int K, int KP) {
    int i = blockIdx.x * blockDim.x + threadIdx.x;
    int total = rows * KP;
    if (i >= total) return;
    int r = i / KP, c = i % KP;
    dst[i] = (c < K) ? (_Float16)src[(size_t)r * K + c] : (_Float16)0.f;
}

// ---------- squared norms of feature rows ----------
template<int CPAD>
__global__ void sqn_kernel(const _Float16* __restrict__ feat, float* __restrict__ sqn) {
    int p = blockIdx.x * blockDim.x + threadIdx.x;
    if (p >= BATCH * NPTS) return;
    const _Float16* f = feat + (size_t)p * CPAD;
    float s = 0.f;
    for (int c = 0; c < CPAD; ++c) { float v = (float)f[c]; s += v * v; }
    sqn[p] = s;
}

// ---------- neg pairwise distance via WMMA: negd = 2*x.xT - |xi|^2 - |xj|^2 ----------
// grid = BATCH*64 blocks (one 16-row tile each), 256 threads (8 waves, 8 col-tiles each)
template<int CPAD>
__global__ void __launch_bounds__(256) dist_kernel(const _Float16* __restrict__ feat,
                                                   const float* __restrict__ sqn,
                                                   float* __restrict__ negd) {
    constexpr int nK = CPAD / 32;
    int blk  = blockIdx.x;
    int b    = blk >> 6;
    int rt   = blk & 63;
    int row0 = rt * 16;
    int lane = threadIdx.x & 31;
    int wv   = threadIdx.x >> 5;
    int m    = lane & 15;
    int half = lane >> 4;

    const _Float16* fb = feat + (size_t)b * NPTS * CPAD;

    // A fragments: rows row0..row0+15, all K chunks (held for the whole wave)
    V16 afrag[nK];
    for (int kc = 0; kc < nK; ++kc) {
        int kb = kc * 32 + half * 8;
        const _Float16* pr = fb + (size_t)(row0 + m) * CPAD + kb;
        afrag[kc].h[0] = *(const h8*)pr;
        afrag[kc].h[1] = *(const h8*)(pr + 16);
    }
    float rn[8];
    for (int e = 0; e < 8; ++e) rn[e] = sqn[b * NPTS + row0 + e + half * 8];

    for (int t = 0; t < 8; ++t) {
        int j0 = (wv * 8 + t) * 16;
        v8f acc = {};
        for (int kc = 0; kc < nK; ++kc) {
            int kb = kc * 32 + half * 8;
            const _Float16* pc = fb + (size_t)(j0 + m) * CPAD + kb;
            V16 bf; bf.h[0] = *(const h8*)pc; bf.h[1] = *(const h8*)(pc + 16);
            acc = __builtin_amdgcn_wmma_f32_16x16x32_f16(false, afrag[kc].v,
                                                         false, bf.v,
                                                         (short)0, acc, false, false);
        }
        float cn = sqn[b * NPTS + j0 + m];
        float* out = negd + ((size_t)(b * NPTS + row0 + half * 8)) * NPTS + j0 + m;
        for (int e = 0; e < 8; ++e)
            out[(size_t)e * NPTS] = 2.f * acc[e] - rn[e] - cn;
    }
}

// ---------- top-20 selection per point (rows live in L2-resident negd) ----------
__global__ void __launch_bounds__(128) topk_kernel(const float* __restrict__ negd,
                                                   int* __restrict__ idx) {
    __shared__ float lv[128][KNN];
    __shared__ int   li[128][KNN];
    int t = threadIdx.x;
    int p = blockIdx.x * 128 + t;
    const float* row = negd + (size_t)p * NPTS;
    int cnt = 0;
    for (int j = 0; j < NPTS; ++j) {
        float v = row[j];
        if (cnt < KNN) {
            int q = cnt++;
            while (q > 0 && lv[t][q - 1] < v) {
                lv[t][q] = lv[t][q - 1]; li[t][q] = li[t][q - 1]; --q;
            }
            lv[t][q] = v; li[t][q] = j;
        } else if (v > lv[t][KNN - 1]) {
            int q = KNN - 1;
            while (q > 0 && lv[t][q - 1] < v) {
                lv[t][q] = lv[t][q - 1]; li[t][q] = li[t][q - 1]; --q;
            }
            lv[t][q] = v; li[t][q] = j;
        }
    }
    for (int q = 0; q < KNN; ++q) idx[(size_t)p * KNN + q] = li[t][q];
}

// ---------- EdgeConv: gather -> (32 x K2) x (K2 x COUT) WMMA -> BN+ELU -> max over k ----------
// 64 threads = 2 waves, one point per wave
template<int CIN, int K2, int COUT>
__global__ void __launch_bounds__(64) edge_kernel(const _Float16* __restrict__ feat, int cstride,
                                                  const int* __restrict__ idx,
                                                  const _Float16* __restrict__ Wf,
                                                  const float* __restrict__ g,
                                                  const float* __restrict__ bb,
                                                  _Float16* __restrict__ out) {
    constexpr int nK = K2 / 32;
    __shared__ alignas(16) _Float16 E[2][32][K2];
    int lane = threadIdx.x & 31;
    int wv   = threadIdx.x >> 5;
    int gp   = blockIdx.x * 2 + wv;
    int b    = gp >> 10;
    int n    = gp & 1023;

    // Build edge rows: m<20 real edges, m>=20 duplicate edge 0 (max-invariant)
    int me  = (lane < KNN) ? lane : 0;
    int nbr = idx[(size_t)gp * KNN + me];
    const _Float16* fc = feat + ((size_t)b * NPTS + n)   * cstride;
    const _Float16* fn = feat + ((size_t)b * NPTS + nbr) * cstride;
    for (int c = 0; c < CIN; ++c) {
        _Float16 ctr = fc[c];
        E[wv][lane][c]       = fn[c] - ctr;
        E[wv][lane][CIN + c] = ctr;
    }
    for (int c = 2 * CIN; c < K2; ++c) E[wv][lane][c] = (_Float16)0.f;
    __syncthreads();

    int m = lane & 15, half = lane >> 4;
    for (int nt = 0; nt < COUT / 16; ++nt) {
        v8f acc0 = {}, acc1 = {};
        for (int kc = 0; kc < nK; ++kc) {
            int kbA = kc * 32 + half * 8;
            V16 a0, a1;
            { const _Float16* pa = &E[wv][m][kbA];
              a0.h[0] = *(const h8*)pa; a0.h[1] = *(const h8*)(pa + 16); }
            { const _Float16* pa = &E[wv][16 + m][kbA];
              a1.h[0] = *(const h8*)pa; a1.h[1] = *(const h8*)(pa + 16); }
            int kbB = kc * 32 + half * 8;
            const _Float16* pb = Wf + (size_t)(nt * 16 + m) * K2 + kbB;
            V16 bf; bf.h[0] = *(const h8*)pb; bf.h[1] = *(const h8*)(pb + 16);
            acc0 = __builtin_amdgcn_wmma_f32_16x16x32_f16(false, a0.v, false, bf.v,
                                                          (short)0, acc0, false, false);
            acc1 = __builtin_amdgcn_wmma_f32_16x16x32_f16(false, a1.v, false, bf.v,
                                                          (short)0, acc1, false, false);
        }
        int oc = nt * 16 + m;
        float sc = g[oc] * BNS, bi = bb[oc];
        float mx = -1e30f;
        for (int e = 0; e < 8; ++e) {
            float h0 = acc0[e] * sc + bi; h0 = h0 > 0.f ? h0 : (__expf(h0) - 1.f);
            float h1 = acc1[e] * sc + bi; h1 = h1 > 0.f ? h1 : (__expf(h1) - 1.f);
            mx = fmaxf(mx, fmaxf(h0, h1));
        }
        mx = fmaxf(mx, __shfl_xor(mx, 16, 32));
        if (lane < 16) out[(size_t)gp * COUT + oc] = (_Float16)mx;
    }
}

// ---------- W5 GEMM over concatenated features, fused BN+ELU + max/mean over N ----------
// 128 threads = 4 waves; wave task = (batch, 16-channel tile of 1024)
__global__ void __launch_bounds__(128) gemm5_kernel(const _Float16* __restrict__ f1,
                                                    const _Float16* __restrict__ f2,
                                                    const _Float16* __restrict__ f3,
                                                    const _Float16* __restrict__ f4,
                                                    const _Float16* __restrict__ W5f,
                                                    const float* __restrict__ g,
                                                    const float* __restrict__ bb,
                                                    float* __restrict__ fout) {
    int lane = threadIdx.x & 31;
    int wv   = threadIdx.x >> 5;
    int task = blockIdx.x * 4 + wv;        // 0..511
    int b    = task >> 6;
    int ct   = task & 63;
    int m = lane & 15, half = lane >> 4;
    int oc = ct * 16 + m;
    float sc = g[oc] * BNS, bi = bb[oc];
    float rmax = -1e30f, rsum = 0.f;

    for (int rt = 0; rt < 64; ++rt) {
        v8f acc = {};
        int row = rt * 16 + m;
        for (int kc = 0; kc < 16; ++kc) {
            V16 a;
            for (int gph = 0; gph < 2; ++gph) {
                int c8 = kc * 32 + half * 8 + gph * 16;
                const _Float16* src; int C, coff;
                if      (c8 <  64) { src = f1; C = 64;  coff = c8; }
                else if (c8 < 128) { src = f2; C = 64;  coff = c8 - 64; }
                else if (c8 < 256) { src = f3; C = 128; coff = c8 - 128; }
                else               { src = f4; C = 256; coff = c8 - 256; }
                a.h[gph] = *(const h8*)(src + ((size_t)(b * NPTS + row)) * C + coff);
            }
            int kbB = kc * 32 + half * 8;
            const _Float16* pb = W5f + (size_t)oc * 512 + kbB;
            V16 bf; bf.h[0] = *(const h8*)pb; bf.h[1] = *(const h8*)(pb + 16);
            acc = __builtin_amdgcn_wmma_f32_16x16x32_f16(false, a.v, false, bf.v,
                                                         (short)0, acc, false, false);
        }
        for (int e = 0; e < 8; ++e) {
            float h = acc[e] * sc + bi; h = h > 0.f ? h : (__expf(h) - 1.f);
            rmax = fmaxf(rmax, h); rsum += h;
        }
    }
    rmax = fmaxf(rmax, __shfl_xor(rmax, 16, 32));
    rsum = rsum + __shfl_xor(rsum, 16, 32);
    if (lane < 16) {
        fout[(size_t)b * 2048 + oc]        = rmax;
        fout[(size_t)b * 2048 + 1024 + oc] = rsum * (1.f / 1024.f);
    }
}

// ---------- small FC layers (fp32 VALU; negligible FLOPs) ----------
__global__ void fc_kernel(const float* __restrict__ in, const float* __restrict__ W,
                          const float* __restrict__ g, const float* __restrict__ bb,
                          float* __restrict__ out, int Bsz, int CIN, int COUT, int mode) {
    int i = blockIdx.x * blockDim.x + threadIdx.x;
    if (i >= Bsz * COUT) return;
    int b = i / COUT, o = i % COUT;
    const float* x = in + (size_t)b * CIN;
    const float* w = W  + (size_t)o * CIN;
    float s = 0.f;
    for (int c = 0; c < CIN; ++c) s += x[c] * w[c];
    if (mode == 0) {
        float h = s * (g[o] * BNS) + bb[o];
        out[i] = h > 0.f ? h : (__expf(h) - 1.f);
    } else {
        out[i] = s + bb[o];
    }
}

extern "C" void kernel_launch(void* const* d_in, const int* in_sizes, int n_in,
                              void* d_out, int out_size, void* d_ws, size_t ws_size,
                              hipStream_t stream) {
    (void)in_sizes; (void)n_in; (void)out_size; (void)ws_size;
    const float* x   = (const float*)d_in[0];
    const float* W1  = (const float*)d_in[1];  const float* g1 = (const float*)d_in[2];  const float* b1 = (const float*)d_in[3];
    const float* W2  = (const float*)d_in[4];  const float* g2 = (const float*)d_in[5];  const float* b2 = (const float*)d_in[6];
    const float* W3  = (const float*)d_in[7];  const float* g3 = (const float*)d_in[8];  const float* b3 = (const float*)d_in[9];
    const float* W4  = (const float*)d_in[10]; const float* g4 = (const float*)d_in[11]; const float* b4 = (const float*)d_in[12];
    const float* W5  = (const float*)d_in[13]; const float* g5 = (const float*)d_in[14]; const float* b5 = (const float*)d_in[15];
    const float* Wl1 = (const float*)d_in[16]; const float* gl1 = (const float*)d_in[17]; const float* bl1 = (const float*)d_in[18];
    const float* Wl2 = (const float*)d_in[19]; const float* gl2 = (const float*)d_in[20]; const float* bl2 = (const float*)d_in[21];
    const float* Wl3 = (const float*)d_in[22]; const float* bl3 = (const float*)d_in[23];
    float* outp = (float*)d_out;

    char* ws = (char*)d_ws;
    size_t off = 0;
    auto take = [&](size_t bytes) -> void* {
        void* p = ws + off;
        off = (off + bytes + 255) & ~(size_t)255;
        return p;
    };
    _Float16* feat0 = (_Float16*)take((size_t)8 * 1024 * 32  * 2);
    _Float16* feat1 = (_Float16*)take((size_t)8 * 1024 * 64  * 2);
    _Float16* feat2 = (_Float16*)take((size_t)8 * 1024 * 64  * 2);
    _Float16* feat3 = (_Float16*)take((size_t)8 * 1024 * 128 * 2);
    _Float16* feat4 = (_Float16*)take((size_t)8 * 1024 * 256 * 2);
    _Float16* W1f   = (_Float16*)take((size_t)64   * 32  * 2);
    _Float16* W2f   = (_Float16*)take((size_t)64   * 128 * 2);
    _Float16* W3f   = (_Float16*)take((size_t)128  * 128 * 2);
    _Float16* W4f   = (_Float16*)take((size_t)256  * 256 * 2);
    _Float16* W5f   = (_Float16*)take((size_t)1024 * 512 * 2);
    float*    sqn   = (float*)take((size_t)8 * 1024 * 4);
    float*    negd  = (float*)take((size_t)8 * 1024 * 1024 * 4);
    int*      idxb  = (int*)take((size_t)8 * 1024 * KNN * 4);
    float*    fbuf  = (float*)take((size_t)8 * 2048 * 4);
    float*    h1b   = (float*)take((size_t)8 * 512 * 4);
    float*    h2b   = (float*)take((size_t)8 * 256 * 4);

    auto cvt = [&](const float* s, _Float16* d, int rows, int K, int KP) {
        int tot = rows * KP;
        cvt_pad_f16<<<(tot + 255) / 256, 256, 0, stream>>>(s, d, rows, K, KP);
    };
    cvt(x,  feat0, 8192, 3,   32);
    cvt(W1, W1f,   64,   6,   32);
    cvt(W2, W2f,   64,   128, 128);
    cvt(W3, W3f,   128,  128, 128);
    cvt(W4, W4f,   256,  256, 256);
    cvt(W5, W5f,   1024, 512, 512);

    // ---- Layer 1 ----
    sqn_kernel<32><<<32, 256, 0, stream>>>(feat0, sqn);
    dist_kernel<32><<<512, 256, 0, stream>>>(feat0, sqn, negd);
    topk_kernel<<<64, 128, 0, stream>>>(negd, idxb);
    edge_kernel<3, 32, 64><<<4096, 64, 0, stream>>>(feat0, 32, idxb, W1f, g1, b1, feat1);
    // ---- Layer 2 ----
    sqn_kernel<64><<<32, 256, 0, stream>>>(feat1, sqn);
    dist_kernel<64><<<512, 256, 0, stream>>>(feat1, sqn, negd);
    topk_kernel<<<64, 128, 0, stream>>>(negd, idxb);
    edge_kernel<64, 128, 64><<<4096, 64, 0, stream>>>(feat1, 64, idxb, W2f, g2, b2, feat2);
    // ---- Layer 3 ----
    sqn_kernel<64><<<32, 256, 0, stream>>>(feat2, sqn);
    dist_kernel<64><<<512, 256, 0, stream>>>(feat2, sqn, negd);
    topk_kernel<<<64, 128, 0, stream>>>(negd, idxb);
    edge_kernel<64, 128, 128><<<4096, 64, 0, stream>>>(feat2, 64, idxb, W3f, g3, b3, feat3);
    // ---- Layer 4 ----
    sqn_kernel<128><<<32, 256, 0, stream>>>(feat3, sqn);
    dist_kernel<128><<<512, 256, 0, stream>>>(feat3, sqn, negd);
    topk_kernel<<<64, 128, 0, stream>>>(negd, idxb);
    edge_kernel<128, 256, 256><<<4096, 64, 0, stream>>>(feat3, 128, idxb, W4f, g4, b4, feat4);

    // ---- W5 + pooling ----
    gemm5_kernel<<<128, 128, 0, stream>>>(feat1, feat2, feat3, feat4, W5f, g5, b5, fbuf);

    // ---- FC head ----
    fc_kernel<<<(8 * 512 + 127) / 128, 128, 0, stream>>>(fbuf, Wl1, gl1, bl1, h1b, 8, 2048, 512, 0);
    fc_kernel<<<(8 * 256 + 127) / 128, 128, 0, stream>>>(h1b,  Wl2, gl2, bl2, h2b, 8, 512,  256, 0);
    fc_kernel<<<(8 * 40  + 127) / 128, 128, 0, stream>>>(h2b,  Wl3, gl2 /*unused*/, bl3, outp, 8, 256, 40, 1);
}
